// UpsampleTransformerV4_53884659695706
// MI455X (gfx1250) — matrix operand order repaired and checked by reading, not compiled, not run on hardware
//
#include <hip/hip_runtime.h>
#include <hip/hip_bf16.h>
#include <math.h>
#include <stdint.h>

typedef _Float16 h16;
typedef unsigned int u32;
typedef unsigned long long u64;
typedef __attribute__((ext_vector_type(16))) _Float16 v16h;
typedef __attribute__((ext_vector_type(8)))  _Float16 v8h;
typedef __attribute__((ext_vector_type(8)))  float    v8f;
typedef __attribute__((ext_vector_type(4)))  u32      v4u;
typedef __attribute__((ext_vector_type(4)))  int      v4i;
typedef __attribute__((ext_vector_type(8)))  int      v8i;

union AFrag { v16h v; v8h p[2]; h16 e[16]; };

#ifndef __has_builtin
#define __has_builtin(x) 0
#endif
#if defined(__HIP_DEVICE_COMPILE__) && __has_builtin(__builtin_amdgcn_tensor_load_to_lds) && __has_builtin(__builtin_amdgcn_s_wait_tensorcnt)
#define HAVE_TDM 1
#else
#define HAVE_TDM 0
#endif

#define CC   128
#define DHD  32
#define KK2  4
#define HU   128
#define WU   192
#define HD_  64
#define WD_  96
#define PIX  32
#define NPIX (2*HU*WU)
#define TPB  128
#define KC   128            /* staged K chunk (halves) */
#define SLOT (16*KC)        /* one wT slot, halves */

/* f16 transposed-weight workspace layout (element offsets) */
#define OFF_Q    0
#define OFF_K    (OFF_Q + 2*128*128)
#define OFF_V    (OFF_K + 2*128*128)
#define OFF_O    (OFF_V + 2*128*128)
#define OFF_FC1  (OFF_O + 2*128*128)     /* [L][512][128] */
#define OFF_FC2  (OFF_FC1 + 2*512*128)   /* [L][128][512] */
#define OFF_OUT1 (OFF_FC2 + 2*128*512)   /* [384][384]    */
#define WS_HALVES (OFF_OUT1 + 384*384)

static constexpr size_t SMEM_BYTES =
    (size_t)(128*128      /* ctx_h  */
           + 128*128      /* cn_h (also q_h, cf) */
           + 32*128       /* x_h    */
           + 32*128       /* bufA (xn/o/h/xf) */
           + 2*SLOT       /* wT double buffer */
           + 128*128      /* k_h (also h1) */
           + 128*128      /* v_h */) * sizeof(h16)
    + 128 * sizeof(float); /* out_acc */

// ---------------- device helpers ----------------

__device__ __forceinline__ v8f vzero8() {
  v8f z = {0.f,0.f,0.f,0.f,0.f,0.f,0.f,0.f};
  return z;
}

__device__ __forceinline__ float gelu_tanh_f(float x) {
  return 0.5f * x * (1.0f + tanhf(0.7978845608028654f * (x + 0.044715f * x * x * x)));
}

// LayerNorm one row of CC f16 values (optionally affine with f32 g/b).
__device__ __forceinline__ void ln_row(const h16* src, h16* dst, float eps,
                                       const float* g, const float* b) {
  float s = 0.f, s2 = 0.f;
  #pragma unroll 8
  for (int i = 0; i < CC; ++i) { float v = (float)src[i]; s += v; s2 += v * v; }
  float m   = s * (1.0f / CC);
  float var = s2 * (1.0f / CC) - m * m;
  float r   = rsqrtf(var + eps);
  #pragma unroll 8
  for (int i = 0; i < CC; ++i) {
    float v = ((float)src[i] - m) * r;
    if (g) v = v * g[i] + b[i];
    dst[i] = (h16)v;
  }
}

#if HAVE_TDM
// TDM: async-load a 16-row x 128-half f16 tile (row stride = K halves) into LDS.
// D# built per CDNA5 ISA ch.8: group0 = count/lds_addr/global_addr/type=2,
// group1 = data_size(2B), tensor dims, tile 128x16, tensor_dim0_stride=K.
__device__ __forceinline__ void tdm_stage(const h16* src, int K, h16* dstLds) {
  u64 ga = (u64)(uintptr_t)src;
  u32 lds = (u32)(uintptr_t)dstLds;     // LDS byte offset (low 32 bits of generic ptr)
  v4u g0;
  g0.x = 1u;                            // count=1, user descriptor
  g0.y = lds;                           // lds_addr
  g0.z = (u32)ga;                       // global_addr[31:0]
  g0.w = (u32)((ga >> 32) & 0x1ffffffull) | (2u << 30);  // addr[56:32] | type=2
  v8i g1;
  g1[0] = (1 << 16);                                        // data_size=1 -> 2 bytes
  g1[1] = (int)(((u32)K & 0xffffu) << 16);                  // tensor_dim0[15:0]
  g1[2] = (int)((((u32)K >> 16) & 0xffffu) | (0xffffu << 16)); // dim0 hi | tensor_dim1 lo
  g1[3] = (int)(128u << 16);                                // tensor_dim1 hi=0 | tile_dim0=128
  g1[4] = 16;                                               // tile_dim1=16, tile_dim2=0
  g1[5] = K;                                                // tensor_dim0_stride[31:0]
  g1[6] = 0;
  g1[7] = 0;
  v4i z4 = {0,0,0,0};
#if __has_include(<hip/amd_detail/amd_gfx1250_TDM.h>)
  v8i z8 = {0,0,0,0,0,0,0,0};
  __builtin_amdgcn_tensor_load_to_lds(g0, g1, z4, z4, z8, 0);
#else
  __builtin_amdgcn_tensor_load_to_lds(g0, g1, z4, z4, 0);
#endif
}
#else
// Fallback: cooperative LDS copy of a 16 x KC f16 tile.
__device__ __forceinline__ void stage_copy(const h16* src, int K, h16* dst, int tid) {
  for (int idx = tid; idx < 256; idx += TPB) {
    int row = idx >> 4, c8 = (idx & 15) << 3;
    *(v8h*)(dst + row*KC + c8) = *(const v8h*)(src + (size_t)row*K + c8);
  }
}
#endif

// K-loop (KC halves) of one 16x16 tile: A row-major in LDS, BT col-major slot.
__device__ __forceinline__ v8f wmma_kloop(const h16* A, int lda, const h16* BT,
                                          v8f c, int lane) {
  const int m    = lane & 15;
  const int sel  = lane >> 4;
  const int aoff = sel * 8;
  const int boff = sel * 8;
  const h16* arow = A + m * lda;
  const h16* bcol = BT + (lane & 15) * KC;
  #pragma unroll
  for (int kb = 0; kb < KC; kb += 32) {
    AFrag a, b;
    a.p[0] = *(const v8h*)(arow + kb + aoff);
    a.p[1] = *(const v8h*)(arow + kb + 16 + aoff);
    b.p[0] = *(const v8h*)(bcol + kb + boff);
    b.p[1] = *(const v8h*)(bcol + kb + 16 + boff);
    c = __builtin_amdgcn_wmma_f32_16x16x32_f16(false, a.v, false, b.v,
                                               (short)0, c, false, false);
  }
  return c;
}

// Dense GEMM: Dst[f16] = act( A[f16] @ WT^T + bias (+ Xres) ), WT is f16 [N][K].
// Stage pipeline: wave0 TDM-issues tile s+1 into the alternate LDS slot, waits
// tensorcnt<=1 for tile s, barrier, all waves WMMA tile s (DMA s+1 overlaps).
__device__ __forceinline__ void gemm_dense(const h16* A, int lda,
                                           const h16* WT, int K, int N,
                                           const float* bias, const h16* Xres, int act,
                                           h16* Dst, int ldd,
                                           h16* wT, int tid, int lane, int wave,
                                           int myT0, int myCnt) {
  const int n    = lane & 15;
  const int half = lane >> 4;
  const int kChunks = K / KC;
  const int nblocks = N >> 4;
  const int S = nblocks * kChunks;
  v8f a0 = vzero8(), a1 = vzero8();
  __syncthreads();                       // previous users of wT slots done
#if HAVE_TDM
  if (wave == 0) tdm_stage(WT, K, wT);   // prologue: stage 0 -> slot 0
#endif
  for (int s = 0; s < S; ++s) {
    const int nb    = s / kChunks;
    const int kcIdx = s - nb * kChunks;
    const int col0  = nb << 4;
    if (kcIdx == 0 && myCnt > 0) {
      float bv = bias ? bias[col0 + n] : 0.f;
      #pragma unroll
      for (int i = 0; i < 8; ++i) {
        float r0 = Xres ? (float)Xres[(size_t)(myT0*16 + i + half*8) * ldd + col0 + n] : 0.f;
        a0[i] = bv + r0;
      }
      if (myCnt > 1) {
        #pragma unroll
        for (int i = 0; i < 8; ++i) {
          float r1 = Xres ? (float)Xres[(size_t)((myT0+1)*16 + i + half*8) * ldd + col0 + n] : 0.f;
          a1[i] = bv + r1;
        }
      }
    }
    __syncthreads();                     // everyone done with slot (s-1)&1
#if HAVE_TDM
    if (wave == 0) {
      if (s + 1 < S) {
        int nb2 = (s + 1) / kChunks;
        int kc2 = ((s + 1) - nb2 * kChunks) * KC;
        tdm_stage(WT + (size_t)(nb2*16)*K + kc2, K, wT + ((s+1)&1)*SLOT);
        __builtin_amdgcn_s_wait_tensorcnt(1);   // tile s landed; s+1 in flight
      } else {
        __builtin_amdgcn_s_wait_tensorcnt(0);
      }
    }
#else
    stage_copy(WT + (size_t)(nb*16)*K + kcIdx*KC, K, wT + (s&1)*SLOT, tid);
#endif
    __syncthreads();                     // publish slot s to all waves
    const h16* wTs = wT + (s & 1) * SLOT;
    if (myCnt > 0) a0 = wmma_kloop(A + (size_t)(myT0*16)*lda + kcIdx*KC, lda, wTs, a0, lane);
    if (myCnt > 1) a1 = wmma_kloop(A + (size_t)((myT0+1)*16)*lda + kcIdx*KC, lda, wTs, a1, lane);
    if (kcIdx == kChunks - 1 && myCnt > 0) {
      #pragma unroll
      for (int i = 0; i < 8; ++i) {
        float v = a0[i]; if (act == 1) v = gelu_tanh_f(v);
        Dst[(size_t)(myT0*16 + i + half*8) * ldd + col0 + n] = (h16)v;
      }
      if (myCnt > 1) {
        #pragma unroll
        for (int i = 0; i < 8; ++i) {
          float v = a1[i]; if (act == 1) v = gelu_tanh_f(v);
          Dst[(size_t)((myT0+1)*16 + i + half*8) * ldd + col0 + n] = (h16)v;
        }
      }
    }
  }
}

// On-the-fly A fragments for fin = [xf, cf, xf-cf]; KC chunks align with the
// 128-wide regions so `region == kcIdx` and is wave-uniform (EXEC stays full).
__device__ __forceinline__ v8f out1_chunk(const h16* xf, const h16* cf, int mt,
                                          const h16* wTs, int region, v8f c, int lane) {
  const int m    = lane & 15;
  const int sel  = lane >> 4;
  const int aoff = sel * 8;
  const int boff = sel * 8;
  const int r    = mt * 16 + m;                 // ctx row (pixel*4 + j)
  const h16* xrow = xf + (r >> 2) * CC;
  const h16* crow = cf + r * CC;
  const h16* bcol = wTs + (lane & 15) * KC;
  #pragma unroll
  for (int kl = 0; kl < KC; kl += 32) {
    AFrag a, b;
    if (region == 0) {
      a.p[0] = *(const v8h*)(xrow + kl + aoff);
      a.p[1] = *(const v8h*)(xrow + kl + 16 + aoff);
    } else if (region == 1) {
      a.p[0] = *(const v8h*)(crow + kl + aoff);
      a.p[1] = *(const v8h*)(crow + kl + 16 + aoff);
    } else {
      AFrag ax, ac;
      ax.p[0] = *(const v8h*)(xrow + kl + aoff);
      ax.p[1] = *(const v8h*)(xrow + kl + 16 + aoff);
      ac.p[0] = *(const v8h*)(crow + kl + aoff);
      ac.p[1] = *(const v8h*)(crow + kl + 16 + aoff);
      #pragma unroll
      for (int e2 = 0; e2 < 16; ++e2) a.e[e2] = ax.e[e2] - ac.e[e2];
    }
    b.p[0] = *(const v8h*)(bcol + kl + boff);
    b.p[1] = *(const v8h*)(bcol + kl + 16 + boff);
    c = __builtin_amdgcn_wmma_f32_16x16x32_f16(false, a.v, false, b.v,
                                               (short)0, c, false, false);
  }
  return c;
}

// ---------------- weight prep: f32 -> f16 transposed [N][K] in workspace ----

__global__ __launch_bounds__(256) void upt_v4_wprep(
    const float* __restrict__ q_w, const float* __restrict__ k_w,
    const float* __restrict__ v_w, const float* __restrict__ o_w,
    const float* __restrict__ fc1_w, const float* __restrict__ fc2_w,
    const float* __restrict__ out1_w, h16* __restrict__ ws) {
  int idx = blockIdx.x * 256 + threadIdx.x;
  if (idx >= WS_HALVES) return;
  float v;
  if (idx < OFF_FC1) {                       // q/k/v/o: [L][128][128] -> T
    int mat = idx >> 15;
    int r = idx & 32767;
    int l = r >> 14, r2 = r & 16383;
    int n = r2 >> 7, k = r2 & 127;
    const float* W = (mat == 0) ? q_w : (mat == 1) ? k_w : (mat == 2) ? v_w : o_w;
    v = W[l*16384 + k*128 + n];
  } else if (idx < OFF_FC2) {                // fc1: [L][128][512] -> [L][512][128]
    int r = idx - OFF_FC1;
    int l = r >> 16, r2 = r & 65535;
    int n = r2 >> 7, k = r2 & 127;
    v = fc1_w[l*65536 + k*512 + n];
  } else if (idx < OFF_OUT1) {               // fc2: [L][512][128] -> [L][128][512]
    int r = idx - OFF_FC2;
    int l = r >> 16, r2 = r & 65535;
    int n = r2 >> 9, k = r2 & 511;
    v = fc2_w[l*65536 + k*128 + n];
  } else {                                   // out1: [384][384] -> T
    int r = idx - OFF_OUT1;
    int n = r / 384, k = r - n*384;
    v = out1_w[k*384 + n];
  }
  ws[idx] = (h16)v;
}

// ---------------- main kernel ----------------

__global__ __launch_bounds__(TPB) void upt_v4_wmma_kernel(
    const float* __restrict__ fm,     const float* __restrict__ fmu,
    const float* __restrict__ ctx_ln_b,
    const float* __restrict__ q_b,    const float* __restrict__ k_b,
    const float* __restrict__ v_b,    const float* __restrict__ o_b,
    const float* __restrict__ fc1_b,  const float* __restrict__ fc2_b,
    const float* __restrict__ out1_b, const float* __restrict__ out2_w,
    const float* __restrict__ out2_b, const float* __restrict__ ctx_ln_g,
    const h16* __restrict__ ws,
    float* __restrict__ out)
{
  extern __shared__ char smem[];
  h16* ctx_h = (h16*)smem;             // [128][128] raw ctx (f16), persistent
  h16* cn_h  = ctx_h + 128*128;        // [128][128] LN(ctx) per layer; later cf
  h16* x_h   = cn_h  + 128*128;        // [32][128]  running x (f16)
  h16* bufA  = x_h   + 32*128;         // [32][128]  xn / o / h / xf
  h16* wT    = bufA  + 32*128;         // 2 slots of [16][KC]
  h16* k_h   = wT    + 2*SLOT;         // [128][128] k   (aliases h1)
  h16* v_h   = k_h   + 128*128;        // [128][128] v
  h16* h1    = k_h;                    // [32][512]  MLP hidden (alias of k/v)
  h16* q_h   = cn_h;                   // [32][128]  q (reuses cn region)
  float* out_acc = (float*)(v_h + 128*128);  // [128]

  const int tid  = threadIdx.x;
  const int lane = tid & 31;
  const int wave = tid >> 5;
  const int pix0 = blockIdx.x * PIX;

  // ---- gather x (from feat_map_up) and ctx window (from feat_map) as f16
  for (int idx = tid; idx < PIX*CC; idx += TPB) {
    int p = idx & (PIX-1), c = idx >> 5;
    int nn = pix0 + p;
    int b  = nn / (HU*WU);
    int rem = nn % (HU*WU);
    int h = rem / WU, w = rem % WU;
    x_h[p*CC + c] = (h16)fmu[(size_t)((b*CC + c)*HU + h)*WU + w];
  }
  for (int idx = tid; idx < PIX*KK2*CC; idx += TPB) {
    int r = idx & (PIX*KK2 - 1), c = idx >> 7;
    int p = r >> 2, j = r & 3;
    int nn = pix0 + p;
    int b  = nn / (HU*WU);
    int rem = nn % (HU*WU);
    int h = rem / WU, w = rem % WU;
    int rh = (h >> 1) + (j >> 1); if (rh > HD_-1) rh = HD_-1;
    int rw = (w >> 1) + (j & 1);  if (rw > WD_-1) rw = WD_-1;
    ctx_h[r*CC + c] = (h16)fm[(size_t)((b*CC + c)*HD_ + rh)*WD_ + rw];
  }
  __syncthreads();

  for (int l = 0; l < 2; ++l) {
    // cn = LN(ctx; eps=1e-5) * g + b   (128 rows, one per thread)
    ln_row(ctx_h + tid*CC, cn_h + tid*CC, 1e-5f, ctx_ln_g + l*CC, ctx_ln_b + l*CC);
    __syncthreads();
    // k = cn @ k_w + k_b ;  v = cn @ v_w + v_b   (8 M-tiles, 2 per wave)
    gemm_dense(cn_h, CC, ws + OFF_K + l*16384, 128, 128, k_b + l*CC, nullptr, 0,
               k_h, CC, wT, tid, lane, wave, wave*2, 2);
    gemm_dense(cn_h, CC, ws + OFF_V + l*16384, 128, 128, v_b + l*CC, nullptr, 0,
               v_h, CC, wT, tid, lane, wave, wave*2, 2);
    __syncthreads();
    // xn = LN(x)
    if (tid < PIX) ln_row(x_h + tid*CC, bufA + tid*CC, 1e-6f, nullptr, nullptr);
    __syncthreads();
    // q = xn @ q_w + q_b  (2 M-tiles, waves 0-1; written into dead cn region)
    gemm_dense(bufA, CC, ws + OFF_Q + l*16384, 128, 128, q_b + l*CC, nullptr, 0,
               q_h, CC, wT, tid, lane, wave, wave, (wave < 2) ? 1 : 0);
    __syncthreads();
    // attention: sim(1x4)/softmax/o per (pixel, head) on VALU -> o into bufA
    {
      int p = tid >> 2, hd = tid & 3;
      int nn = pix0 + p;
      int rem = nn % (HU*WU);
      int h = rem / WU, w = rem % WU;
      float slope = powf(24.0f, -0.25f * (float)(hd + 1));
      const h16* qp = q_h + p*CC + hd*DHD;
      float simv[4], e4[4];
      float mx = -1e30f;
      #pragma unroll
      for (int j = 0; j < 4; ++j) {
        int rh = (h >> 1) + (j >> 1); if (rh > HD_-1) rh = HD_-1;
        int rw = (w >> 1) + (j & 1);  if (rw > WD_-1) rw = WD_-1;
        float cd = -fabsf((float)(h - 2*rh)) - fabsf((float)(w - 2*rw));
        const h16* kp = k_h + (p*4 + j)*CC + hd*DHD;
        float dot = 0.f;
        #pragma unroll
        for (int d = 0; d < DHD; ++d) dot += (float)qp[d] * (float)kp[d];
        simv[j] = dot * 0.17677669529663687f + slope * cd;  // 1/sqrt(32)
        mx = fmaxf(mx, simv[j]);
      }
      float ssum = 0.f;
      #pragma unroll
      for (int j = 0; j < 4; ++j) { e4[j] = __expf(simv[j] - mx); ssum += e4[j]; }
      float inv = 1.f / ssum;
      #pragma unroll 4
      for (int d = 0; d < DHD; ++d) {
        float o = 0.f;
        #pragma unroll
        for (int j = 0; j < 4; ++j) o += e4[j] * (float)v_h[(p*4 + j)*CC + hd*DHD + d];
        bufA[p*CC + hd*DHD + d] = (h16)(o * inv);
      }
    }
    __syncthreads();
    // x = x + o @ o_w + o_b   (residual folded into C-init)
    gemm_dense(bufA, CC, ws + OFF_O + l*16384, 128, 128, o_b + l*CC, x_h, 0,
               x_h, CC, wT, tid, lane, wave, wave, (wave < 2) ? 1 : 0);
    __syncthreads();
    // h = LN(x)
    if (tid < PIX) ln_row(x_h + tid*CC, bufA + tid*CC, 1e-6f, nullptr, nullptr);
    __syncthreads();
    // h1 = gelu_tanh(h @ fc1 + b)   (N = 512)
    gemm_dense(bufA, CC, ws + OFF_FC1 + l*65536, 128, 512, fc1_b + l*512,
               nullptr, 1, h1, 512, wT, tid, lane, wave, wave, (wave < 2) ? 1 : 0);
    __syncthreads();
    // x = x + h1 @ fc2 + b   (K = 512, four staged chunks)
    gemm_dense(h1, 512, ws + OFF_FC2 + l*65536, 512, 128, fc2_b + l*CC, x_h, 0,
               x_h, CC, wT, tid, lane, wave, wave, (wave < 2) ? 1 : 0);
    __syncthreads();
  }

  // final LNs: xf -> bufA (32 rows), cf -> cn_h (128 rows)
  if (tid < PIX) ln_row(x_h + tid*CC, bufA + tid*CC, 1e-6f, nullptr, nullptr);
  ln_row(ctx_h + tid*CC, cn_h + tid*CC, 1e-6f, nullptr, nullptr);
  out_acc[tid] = out2_b[0];
  __syncthreads();

  // fused: out_acc[r] += sum_n gelu_erf(fin@out1_w + out1_b)[r][n] * out2_w[n]
  {
    const int n    = lane & 15;
    const int half = lane >> 4;
    const h16* WT = ws + OFF_OUT1;     // [384][384] f16, row stride 384
    v8f a0 = vzero8(), a1 = vzero8();
    __syncthreads();
#if HAVE_TDM
    if (wave == 0) tdm_stage(WT, 384, wT);
#endif
    for (int s = 0; s < 72; ++s) {     // 24 N-blocks x 3 K-chunks
      const int nb    = s / 3;
      const int kcIdx = s - nb * 3;
      const int col0  = nb << 4;
      if (kcIdx == 0) { a0 = vzero8(); a1 = vzero8(); }
      __syncthreads();
#if HAVE_TDM
      if (wave == 0) {
        if (s + 1 < 72) {
          int nb2 = (s + 1) / 3;
          int kc2 = ((s + 1) - nb2 * 3) * KC;
          tdm_stage(WT + (size_t)(nb2*16)*384 + kc2, 384, wT + ((s+1)&1)*SLOT);
          __builtin_amdgcn_s_wait_tensorcnt(1);
        } else {
          __builtin_amdgcn_s_wait_tensorcnt(0);
        }
      }
#else
      stage_copy(WT + (size_t)(nb*16)*384 + kcIdx*KC, 384, wT + (s&1)*SLOT, tid);
#endif
      __syncthreads();
      const h16* wTs = wT + (s & 1) * SLOT;
      a0 = out1_chunk(bufA, cn_h, wave*2,     wTs, kcIdx, a0, lane);
      a1 = out1_chunk(bufA, cn_h, wave*2 + 1, wTs, kcIdx, a1, lane);
      if (kcIdx == 2) {
        float w2 = out2_w[col0 + n];
        float b1 = out1_b[col0 + n];
        #pragma unroll
        for (int i = 0; i < 8; ++i) {
          float v0 = a0[i] + b1;
          float g0 = 0.5f * v0 * (1.0f + erff(v0 * 0.7071067811865475f));
          atomicAdd(&out_acc[(wave*2)*16 + i + half*8], g0 * w2);
          float v1 = a1[i] + b1;
          float g1 = 0.5f * v1 * (1.0f + erff(v1 * 0.7071067811865475f));
          atomicAdd(&out_acc[(wave*2 + 1)*16 + i + half*8], g1 * w2);
        }
      }
    }
  }
  __syncthreads();

  // softmax over K2 = 4 and scatter to (B, K2, Hu, Wu)
  if (tid < PIX) {
    int p = tid;
    int nn = pix0 + p;
    int b  = nn / (HU*WU);
    int rem = nn % (HU*WU);
    int h = rem / WU, w = rem % WU;
    float s0 = out_acc[p*4+0], s1 = out_acc[p*4+1];
    float s2 = out_acc[p*4+2], s3 = out_acc[p*4+3];
    float mx = fmaxf(fmaxf(s0, s1), fmaxf(s2, s3));
    float e0 = __expf(s0-mx), e1 = __expf(s1-mx), e2 = __expf(s2-mx), e3 = __expf(s3-mx);
    float inv = 1.f / (e0 + e1 + e2 + e3);
    size_t base = (size_t)b * KK2 * HU * WU + (size_t)h * WU + w;
    const size_t plane = (size_t)HU * WU;
    out[base + 0*plane] = e0 * inv;
    out[base + 1*plane] = e1 * inv;
    out[base + 2*plane] = e2 * inv;
    out[base + 3*plane] = e3 * inv;
  }
}

// ---------------- launcher ----------------

extern "C" void kernel_launch(void* const* d_in, const int* in_sizes, int n_in,
                              void* d_out, int out_size, void* d_ws, size_t ws_size,
                              hipStream_t stream) {
  (void)in_sizes; (void)n_in; (void)out_size; (void)ws_size;
  const float* fm       = (const float*)d_in[0];
  const float* fmu      = (const float*)d_in[1];
  const float* ctx_ln_b = (const float*)d_in[2];
  const float* q_w      = (const float*)d_in[3];
  const float* q_b      = (const float*)d_in[4];
  const float* k_w      = (const float*)d_in[5];
  const float* k_b      = (const float*)d_in[6];
  const float* v_w      = (const float*)d_in[7];
  const float* v_b      = (const float*)d_in[8];
  const float* o_w      = (const float*)d_in[9];
  const float* o_b      = (const float*)d_in[10];
  const float* fc1_w    = (const float*)d_in[11];
  const float* fc1_b    = (const float*)d_in[12];
  const float* fc2_w    = (const float*)d_in[13];
  const float* fc2_b    = (const float*)d_in[14];
  const float* out1_w   = (const float*)d_in[15];
  const float* out1_b   = (const float*)d_in[16];
  const float* out2_w   = (const float*)d_in[17];
  const float* out2_b   = (const float*)d_in[18];
  const float* ctx_ln_g = (const float*)d_in[19];
  float* out = (float*)d_out;
  h16* ws = (h16*)d_ws;

  // 1) one-time (per launch) weight convert+transpose into workspace (f16)
  upt_v4_wprep<<<dim3((WS_HALVES + 255) / 256), dim3(256), 0, stream>>>(
      q_w, k_w, v_w, o_w, fc1_w, fc2_w, out1_w, ws);

  // 2) fused transformer upsampler
  upt_v4_wmma_kernel<<<dim3(NPIX / PIX), dim3(TPB), SMEM_BYTES, stream>>>(
      fm, fmu, ctx_ln_b, q_b, k_b, v_b, o_b, fc1_b, fc2_b,
      out1_b, out2_w, out2_b, ctx_ln_g, ws, out);
}
